// DeepSetStrategyModel_30365418782904
// MI455X (gfx1250) — matrix operand order repaired and checked by reading, not compiled, not run on hardware
//
#include <hip/hip_runtime.h>
#include <hip/hip_bf16.h>

#define E_EDGES 2000000
#define N_NODES 50000

typedef __attribute__((ext_vector_type(16))) _Float16 v16h;
typedef __attribute__((ext_vector_type(8)))  _Float16 v8h;
typedef __attribute__((ext_vector_type(8)))  float    v8f;
typedef unsigned int u32x4 __attribute__((ext_vector_type(4)));
typedef int          i32x8 __attribute__((ext_vector_type(8)));
typedef int          i32x4 __attribute__((ext_vector_type(4)));

__device__ __forceinline__ float lrelu(float x) { return x >= 0.f ? x : 0.01f * x; }

__device__ __forceinline__ v8f wmma16(v16h a, v16h b, v8f c) {
  // D = A(16x32 f16) * B(32x16 f16) + C(16x16 f32)
  return __builtin_amdgcn_wmma_f32_16x16x32_f16(false, a, false, b, (short)0, c, false, false);
}

__device__ __forceinline__ v16h combine8(v8h lo, v8h hi) {
  v16h a;
#pragma unroll
  for (int i = 0; i < 8; ++i) { a[i] = lo[i]; a[8 + i] = hi[i]; }
  return a;
}

// A fragment from a row-major f16 [*,32] row.
// ISA layout (16-bit A 16x32): lanes 0-15 row m hold K{0..7,16..23}; lanes 16-31 hold K{8..15,24..31}.
__device__ __forceinline__ v16h load_a_h16(const _Float16* row, int hi) {
  v8h lo = *(const v8h*)(row + (hi ? 8 : 0));
  v8h h2 = *(const v8h*)(row + (hi ? 24 : 16));
  return combine8(lo, h2);
}

__device__ __forceinline__ v16h load_a_f32(const float* row, int hi) {
  v16h a;
  int o1 = hi ? 8 : 0, o2 = hi ? 24 : 16;
#pragma unroll
  for (int i = 0; i < 8; ++i) {
    a[i]     = (_Float16)row[o1 + i];
    a[8 + i] = (_Float16)row[o2 + i];
  }
  return a;
}

// B fragment from packed buffer (global or LDS): [frag][lane][16] f16 contiguous.
__device__ __forceinline__ v16h load_b_pk(const _Float16* pk, int frag, int lane) {
  const _Float16* p = pk + (((size_t)frag * 32 + lane) << 4);
  v8h lo = *(const v8h*)p;
  v8h hi = *(const v8h*)(p + 8);
  return combine8(lo, hi);
}

__device__ __forceinline__ void atomicMaxFloat(float* addr, float val) {
  if (val >= 0.f) atomicMax((int*)addr, __float_as_int(val));
  else            atomicMin((unsigned int*)addr, __float_as_uint(val));
}

// ---- Tensor Data Mover: stage a 1-D run of packed f16 weights into LDS ----
// D# per CDNA5 ISA ch.8: group0 = {count|flags, lds_addr, global_addr lo/hi+type},
// group1 = {data_size, tensor_dim0, tile_dim0, tile_dim1=0 (1-D), strides}.
__device__ __forceinline__ void tdm_stage_weights(const _Float16* gsrc, _Float16* lds_dst,
                                                  unsigned nelems) {
  if (threadIdx.x < 32) {   // one wave issues the DMA; EXEC is ignored by TDM
    unsigned lds_off = (unsigned)(uintptr_t)lds_dst;       // low 32 bits = LDS byte address
    unsigned long long ga = (unsigned long long)(uintptr_t)gsrc;
    u32x4 g0;
    g0[0] = 1u;                                            // count=1 (valid user descriptor)
    g0[1] = lds_off;                                       // lds_addr
    g0[2] = (unsigned)ga;                                  // global_addr[31:0]
    g0[3] = (unsigned)((ga >> 32) & 0x01ffffffu) | (2u << 30); // global_addr[56:32] | type=2
    i32x8 g1;
    g1[0] = 1 << 16;                                       // data_size=1 -> 2 bytes/elem
    g1[1] = (int)((nelems & 0xffffu) << 16);               // tensor_dim0[15:0]
    g1[2] = (int)(nelems >> 16);                           // tensor_dim0[31:16], tensor_dim1=0
    g1[3] = (int)((nelems & 0xffffu) << 16);               // tile_dim0 = nelems
    g1[4] = 0;                                             // tile_dim1=0 (1-D), tile_dim2=0
    g1[5] = (int)nelems;                                   // tensor_dim0_stride[31:0]
    g1[6] = 0;
    g1[7] = 0;
    i32x4 z4 = {0, 0, 0, 0};                               // groups 2/3 unused (<=2-D tensor)
    i32x8 z8 = {0, 0, 0, 0, 0, 0, 0, 0};
    __builtin_amdgcn_tensor_load_to_lds(g0, g1, z4, z4, z8, 0);
    __builtin_amdgcn_s_wait_tensorcnt(0);
  }
  __syncthreads();
}

// ---------------- weight packing into B-fragment layout ----------------
// Region-contiguous frag map (frag = 512 f16 = 1KB):
//  [0..3]   edge_t l0: tw1(2) tw2(2)       [4..7]   edge_t l1
//  [8..13]  edge_u l0: uw1(4) uw2(2)       [14..19] edge_u l1
//  [20..25] head:      hw1(4) hw2(2)                      Total 26 frags.
__global__ void pack_weights_kernel(const float* __restrict__ tw1, const float* __restrict__ tw2,
                                    const float* __restrict__ uw1, const float* __restrict__ uw2,
                                    const float* __restrict__ hw1, const float* __restrict__ hw2,
                                    _Float16* __restrict__ pk) {
  int t = blockIdx.x * blockDim.x + threadIdx.x;
  if (t >= 26 * 512) return;
  int f = t >> 9;
  int lane = (t >> 4) & 31;
  int j = t & 15;
  const float* W; int fb;
  if      (f < 2)  { W = tw1;        fb = 0;  }
  else if (f < 4)  { W = tw2;        fb = 2;  }
  else if (f < 6)  { W = tw1 + 1024; fb = 4;  }
  else if (f < 8)  { W = tw2 + 1024; fb = 6;  }
  else if (f < 12) { W = uw1;        fb = 8;  }
  else if (f < 14) { W = uw2;        fb = 12; }
  else if (f < 18) { W = uw1 + 2048; fb = 14; }
  else if (f < 20) { W = uw2 + 1024; fb = 18; }
  else if (f < 24) { W = hw1;        fb = 20; }
  else             { W = hw2;        fb = 24; }
  int lf = f - fb;
  int kc = lf >> 1, nt = lf & 1;           // K-chunk of 32, N-tile of 16
  int col = nt * 16 + (lane & 15);
  int k = kc * 32 + ((j & 8) << 1) + ((lane & 16) ? 8 : 0) + (j & 7);
  pk[t] = (_Float16)W[k * 32 + col];
}

// ---------------- h0 = edge_attr * w_in + b_in (f16) ----------------
__global__ void h0_kernel(const float* __restrict__ ea, const float* __restrict__ w_in,
                          const float* __restrict__ b_in, _Float16* __restrict__ h16) {
  long t = (long)blockIdx.x * blockDim.x + threadIdx.x;
  if (t >= (long)E_EDGES * 32) return;
  long e = t >> 5; int d = (int)(t & 31);
  h16[t] = (_Float16)(ea[e] * w_in[d] + b_in[d]);
}

__global__ void zero_kernel(float* __restrict__ p, int n) {
  int i = blockIdx.x * blockDim.x + threadIdx.x;
  if (i < n) p[i] = 0.f;
}

__global__ void init_md_kernel(float* __restrict__ m, float* __restrict__ d) {
  int i = blockIdx.x * blockDim.x + threadIdx.x;
  if (i < N_NODES) { m[i] = __int_as_float(0xFF800000); d[i] = 0.f; }
}

// ---------------- t = lrelu(h@tw1+tb1)@tw2+tb2 ; atomic scatter into s ----------------
__global__ void __launch_bounds__(256) edge_t_kernel(
    const _Float16* __restrict__ h16, const int* __restrict__ idx,
    const _Float16* __restrict__ pk_region,           // 4 frags: tw1 n0,n1 | tw2 n0,n1
    const float* __restrict__ tb1, const float* __restrict__ tb2, float* __restrict__ s) {
  __shared__ _Float16 xp[8][16][32];
  __shared__ _Float16 wl[4 * 512];
  tdm_stage_weights(pk_region, wl, 4 * 512);

  int wave = threadIdx.x >> 5, lane = threadIdx.x & 31;
  int m = lane & 15, hi = lane >> 4;
  long base = ((long)blockIdx.x * 8 + wave) * 16;

  v16h a = load_a_h16(h16 + (base + m) * 32, hi);
  v8f c0 = {}; v8f c1 = {};
  c0 = wmma16(a, load_b_pk(wl, 0, lane), c0);
  c1 = wmma16(a, load_b_pk(wl, 1, lane), c1);
  int col0 = m, col1 = 16 + m;
  float b0 = tb1[col0], b1 = tb1[col1];
#pragma unroll
  for (int v = 0; v < 8; ++v) {        // D layout: lane -> column, VGPR -> row
    int row = v + hi * 8;
    xp[wave][row][col0] = (_Float16)lrelu(c0[v] + b0);
    xp[wave][row][col1] = (_Float16)lrelu(c1[v] + b1);
  }
  __syncthreads();
  v16h az = load_a_h16(&xp[wave][m][0], hi);
  v8f t0 = {}; v8f t1 = {};
  t0 = wmma16(az, load_b_pk(wl, 2, lane), t0);
  t1 = wmma16(az, load_b_pk(wl, 3, lane), t1);
  float o0 = tb2[col0], o1 = tb2[col1];
#pragma unroll
  for (int v = 0; v < 8; ++v) {
    int row = v + hi * 8;
    int node = idx[base + row];
    atomicAdd(&s[(long)node * 32 + col0], t0[v] + o0);
    atomicAdd(&s[(long)node * 32 + col1], t1[v] + o1);
  }
}

// ---------------- h = lrelu([h | s[idx]]@uw1+ub1)@uw2+ub2 ----------------
__global__ void __launch_bounds__(256) edge_u_kernel(
    _Float16* __restrict__ h16, const int* __restrict__ idx, const float* __restrict__ s,
    const _Float16* __restrict__ pk_region,           // 6 frags: uw1 k0n0,k0n1,k1n0,k1n1 | uw2 n0,n1
    const float* __restrict__ ub1, const float* __restrict__ ub2) {
  __shared__ _Float16 xp[8][16][32];
  __shared__ _Float16 wl[6 * 512];
  tdm_stage_weights(pk_region, wl, 6 * 512);

  int wave = threadIdx.x >> 5, lane = threadIdx.x & 31;
  int m = lane & 15, hi = lane >> 4;
  long base = ((long)blockIdx.x * 8 + wave) * 16;

  v16h ah = load_a_h16(h16 + (base + m) * 32, hi);
  int node = idx[base + m];
  v16h as = load_a_f32(s + (long)node * 32, hi);
  v8f c0 = {}; v8f c1 = {};
  c0 = wmma16(ah, load_b_pk(wl, 0, lane), c0);   // K 0..31,  N-tile 0
  c1 = wmma16(ah, load_b_pk(wl, 1, lane), c1);   // K 0..31,  N-tile 1
  c0 = wmma16(as, load_b_pk(wl, 2, lane), c0);   // K 32..63, N-tile 0
  c1 = wmma16(as, load_b_pk(wl, 3, lane), c1);   // K 32..63, N-tile 1
  int col0 = m, col1 = 16 + m;
  float b0 = ub1[col0], b1 = ub1[col1];
#pragma unroll
  for (int v = 0; v < 8; ++v) {
    int row = v + hi * 8;
    xp[wave][row][col0] = (_Float16)lrelu(c0[v] + b0);
    xp[wave][row][col1] = (_Float16)lrelu(c1[v] + b1);
  }
  __syncthreads();
  v16h az = load_a_h16(&xp[wave][m][0], hi);
  v8f t0 = {}; v8f t1 = {};
  t0 = wmma16(az, load_b_pk(wl, 4, lane), t0);
  t1 = wmma16(az, load_b_pk(wl, 5, lane), t1);
  float o0 = ub2[col0], o1 = ub2[col1];
#pragma unroll
  for (int v = 0; v < 8; ++v) {
    int row = v + hi * 8;
    h16[(base + row) * 32 + col0] = (_Float16)(t0[v] + o0);
    h16[(base + row) * 32 + col1] = (_Float16)(t1[v] + o1);
  }
}

// ---------------- head MLP -> values, atomic segment max ----------------
__global__ void __launch_bounds__(256) head_kernel(
    const float* __restrict__ ea, const _Float16* __restrict__ h16, const int* __restrict__ idx,
    const float* __restrict__ w_in, const float* __restrict__ b_in,
    const _Float16* __restrict__ pk_region,           // 6 frags: hw1(4) | hw2(2)
    const float* __restrict__ hb1, const float* __restrict__ hb2,
    const float* __restrict__ hw3, const float* __restrict__ hb3,
    float* __restrict__ values, float* __restrict__ mmax) {
  __shared__ _Float16 xp[8][16][32];
  __shared__ float    vt[8][16][32];
  __shared__ _Float16 wl[6 * 512];
  tdm_stage_weights(pk_region, wl, 6 * 512);

  int wave = threadIdx.x >> 5, lane = threadIdx.x & 31;
  int m = lane & 15, hi = lane >> 4;
  long base = ((long)blockIdx.x * 8 + wave) * 16;

  // recompute h0 fragment on the fly (rank-1: ea[e]*w_in + b_in)
  float eav = ea[base + m];
  int o1 = hi ? 8 : 0, o2 = hi ? 24 : 16;
  v16h a0;
#pragma unroll
  for (int i = 0; i < 8; ++i) {
    a0[i]     = (_Float16)(eav * w_in[o1 + i] + b_in[o1 + i]);
    a0[8 + i] = (_Float16)(eav * w_in[o2 + i] + b_in[o2 + i]);
  }
  v16h ah = load_a_h16(h16 + (base + m) * 32, hi);
  v8f c0 = {}; v8f c1 = {};
  c0 = wmma16(a0, load_b_pk(wl, 0, lane), c0);
  c1 = wmma16(a0, load_b_pk(wl, 1, lane), c1);
  c0 = wmma16(ah, load_b_pk(wl, 2, lane), c0);
  c1 = wmma16(ah, load_b_pk(wl, 3, lane), c1);
  int col0 = m, col1 = 16 + m;
  float b0 = hb1[col0], b1 = hb1[col1];
#pragma unroll
  for (int v = 0; v < 8; ++v) {
    int row = v + hi * 8;
    xp[wave][row][col0] = (_Float16)lrelu(c0[v] + b0);
    xp[wave][row][col1] = (_Float16)lrelu(c1[v] + b1);
  }
  __syncthreads();
  v16h az = load_a_h16(&xp[wave][m][0], hi);
  v8f t0 = {}; v8f t1 = {};
  t0 = wmma16(az, load_b_pk(wl, 4, lane), t0);
  t1 = wmma16(az, load_b_pk(wl, 5, lane), t1);
  float o0 = hb2[col0], o12 = hb2[col1];
#pragma unroll
  for (int v = 0; v < 8; ++v) {
    int row = v + hi * 8;
    vt[wave][row][col0] = lrelu(t0[v] + o0);
    vt[wave][row][col1] = lrelu(t1[v] + o12);
  }
  __syncthreads();
  if (lane < 16) {
    int row = lane;
    float acc = hb3[0];
#pragma unroll
    for (int c = 0; c < 32; ++c) acc += vt[wave][row][c] * hw3[c];
    long e = base + row;
    values[e] = acc;
    atomicMaxFloat(&mmax[idx[e]], acc);
  }
}

// ---------------- softmax numerator + denominator, then normalize ----------------
__global__ void exp_kernel(const float* __restrict__ values, const int* __restrict__ idx,
                           const float* __restrict__ mmax, float* __restrict__ out,
                           float* __restrict__ dsum) {
  long e = (long)blockIdx.x * blockDim.x + threadIdx.x;
  if (e >= E_EDGES) return;
  float x = __expf(values[e] - mmax[idx[e]]);
  out[e] = x;
  atomicAdd(&dsum[idx[e]], x);
}

__global__ void div_kernel(float* __restrict__ out, const int* __restrict__ idx,
                           const float* __restrict__ dsum) {
  long e = (long)blockIdx.x * blockDim.x + threadIdx.x;
  if (e >= E_EDGES) return;
  out[e] = out[e] / dsum[idx[e]];
}

extern "C" void kernel_launch(void* const* d_in, const int* in_sizes, int n_in,
                              void* d_out, int out_size, void* d_ws, size_t ws_size,
                              hipStream_t stream) {
  const float* edge_attr  = (const float*)d_in[0];
  const int*   edge_index = (const int*)d_in[1];
  const float* w_in = (const float*)d_in[2];
  const float* b_in = (const float*)d_in[3];
  const float* tw1  = (const float*)d_in[4];
  const float* tb1  = (const float*)d_in[5];
  const float* tw2  = (const float*)d_in[6];
  const float* tb2  = (const float*)d_in[7];
  const float* uw1  = (const float*)d_in[8];
  const float* ub1  = (const float*)d_in[9];
  const float* uw2  = (const float*)d_in[10];
  const float* ub2  = (const float*)d_in[11];
  const float* hw1  = (const float*)d_in[12];
  const float* hb1  = (const float*)d_in[13];
  const float* hw2  = (const float*)d_in[14];
  const float* hb2  = (const float*)d_in[15];
  const float* hw3  = (const float*)d_in[16];
  const float* hb3  = (const float*)d_in[17];
  (void)in_sizes; (void)n_in; (void)out_size; (void)ws_size;
  const int* idx = edge_index;            // edge_index[0] = first E entries
  float* out = (float*)d_out;

  char* w = (char*)d_ws;
  size_t off = 0;
  auto alloc = [&](size_t bytes) -> void* {
    void* p = w + off;
    off += (bytes + 255) & ~(size_t)255;
    return p;
  };
  _Float16* h16   = (_Float16*)alloc((size_t)E_EDGES * 32 * 2); // 128 MB
  float*    s     = (float*)   alloc((size_t)N_NODES * 32 * 4); // 6.4 MB
  float*    mmax  = (float*)   alloc((size_t)N_NODES * 4);
  float*    dsum  = (float*)   alloc((size_t)N_NODES * 4);
  float*    vals  = (float*)   alloc((size_t)E_EDGES * 4);      // 8 MB
  _Float16* pk    = (_Float16*)alloc((size_t)26 * 512 * 2);     // 26 KB packed weights

  pack_weights_kernel<<<52, 256, 0, stream>>>(tw1, tw2, uw1, uw2, hw1, hw2, pk);
  h0_kernel<<<(E_EDGES * 32) / 256, 256, 0, stream>>>(edge_attr, w_in, b_in, h16);

  const int EB = E_EDGES / 128;  // 15625 blocks, 8 waves x 16 edges each
  for (int l = 0; l < 2; ++l) {
    zero_kernel<<<(N_NODES * 32 + 255) / 256, 256, 0, stream>>>(s, N_NODES * 32);
    edge_t_kernel<<<EB, 256, 0, stream>>>(h16, idx,
        pk + (size_t)(l * 4) * 512, tb1 + l * 32, tb2 + l * 32, s);
    edge_u_kernel<<<EB, 256, 0, stream>>>(h16, idx, s,
        pk + (size_t)(8 + l * 6) * 512, ub1 + l * 32, ub2 + l * 32);
  }

  init_md_kernel<<<(N_NODES + 255) / 256, 256, 0, stream>>>(mmax, dsum);
  head_kernel<<<EB, 256, 0, stream>>>(edge_attr, h16, idx, w_in, b_in,
      pk + (size_t)20 * 512, hb1, hb2, hw3, hb3, vals, mmax);
  exp_kernel<<<(E_EDGES + 255) / 256, 256, 0, stream>>>(vals, idx, mmax, out, dsum);
  div_kernel<<<(E_EDGES + 255) / 256, 256, 0, stream>>>(out, idx, dsum);
}
